// AttentionHead_82592221102450
// MI455X (gfx1250) — matrix operand order repaired and checked by reading, not compiled, not run on hardware
//
#include <hip/hip_runtime.h>
#include <hip/hip_bf16.h>

// Flash-attention single head on gfx1250 (CDNA5), bf16 WMMA pipeline.
// Phase 0: transpose weights fp32[768][64] -> bf16 Wt[64][768].
// Phase 1: QKV projection GEMMs (v_wmma_f32_16x16x32_bf16), x tile staged in
//          LDS; writes Q,K row-major bf16 and V transposed bf16.
// Phase 2: flash attention computed TRANSPOSED: S^T = K Q^T and O^T = V^T P^T,
//          so each lane owns exactly one query column -> per-lane scalar
//          softmax state (1 shfl per reduction), packed ds_store_b128 for the
//          P^T relayout, contiguous 32B output stores.

typedef __bf16 bf16;
typedef __attribute__((ext_vector_type(16))) bf16  v16bf;
typedef __attribute__((ext_vector_type(8)))  bf16  v8bf;
typedef __attribute__((ext_vector_type(8)))  float v8f;

#define BATCH   8
#define SEQ     2048
#define DMODEL  768
#define HEAD    64
#define QTILES_PER_B (SEQ / 16)   // 128

static __device__ inline v16bf ld_v16(const bf16* p) {
    return *(const v16bf*)p;
}
static __device__ inline v8bf ld_v8(const bf16* p) {
    return *(const v8bf*)p;
}
static __device__ inline v16bf combine8(v8bf lo, v8bf hi) {
    v16bf r;
#pragma unroll
    for (int e = 0; e < 8; ++e) { r[e] = lo[e]; r[e + 8] = hi[e]; }
    return r;
}

#define WMMA_BF16(A, B, C) \
    __builtin_amdgcn_wmma_f32_16x16x32_bf16(false, (A), false, (B), (short)0, (C), false, false)

// ---------------------------------------------------------------------------
// Phase 0: weight transpose + bf16 convert.  W[768][64] f32 -> Wt[64][768] bf16.
// ---------------------------------------------------------------------------
__global__ void __launch_bounds__(256)
attn_wt_transpose_kernel(const float* __restrict__ Wq,
                         const float* __restrict__ Wk,
                         const float* __restrict__ Wv,
                         bf16* __restrict__ Wtq, bf16* __restrict__ Wtk,
                         bf16* __restrict__ Wtv) {
    const int idx = blockIdx.x * 256 + threadIdx.x;   // 0 .. 768*64-1
    const int d = idx >> 6;        // row in W   (model dim)
    const int h = idx & 63;        // col in W   (head dim)
    const size_t t = (size_t)h * DMODEL + d;
    Wtq[t] = (bf16)Wq[idx];
    Wtk[t] = (bf16)Wk[idx];
    Wtv[t] = (bf16)Wv[idx];
}

// ---------------------------------------------------------------------------
// Phase 1: Q/K/V projections.  One block per 16-row tile of [B*S, D].
// Block = 128 threads = 4 waves; wave w computes output columns w*16..w*16+15
// for all three weight matrices.  The 16x32 x tile is staged in LDS (bf16)
// once per k-step; B operands are contiguous v16bf loads from the transposed
// bf16 weights (L2 resident: 3 x 96 KB).
// ---------------------------------------------------------------------------
__global__ void __launch_bounds__(128)
attn_qkv_proj_kernel(const float* __restrict__ x,
                     const bf16* __restrict__ Wtq, const float* __restrict__ bq,
                     const bf16* __restrict__ Wtk, const float* __restrict__ bk,
                     const bf16* __restrict__ Wtv, const float* __restrict__ bv,
                     bf16* __restrict__ Qb, bf16* __restrict__ Kb,
                     bf16* __restrict__ Vt) {
    __shared__ bf16 Xlds[16 * 32];        // current 16x32 x tile, bf16

    const int lane = threadIdx.x & 31;
    const int hi   = lane >> 4;           // half-wave select
    const int col  = lane & 15;

    const int tile = blockIdx.x;          // 0 .. B*S/16-1
    const int b    = tile >> 7;           // /128
    const int s0   = (tile & 127) << 4;
    const int ga   = b * SEQ + s0;        // global row base
    const int h    = ((threadIdx.x >> 5) << 4) + col;  // output column (N-dim)

    // staging assignment: thread t loads 4 consecutive floats of row (t>>3)
    const int str = threadIdx.x >> 3;
    const int stc = (threadIdx.x & 7) << 2;

    v8f accQ = {}, accK = {}, accV = {};

    for (int kb = 0; kb < DMODEL; kb += 32) {
        // ---- cooperative stage of x tile into LDS (coalesced 16B loads) ----
        {
            const float4 xv =
                *(const float4*)(x + (size_t)(ga + str) * DMODEL + kb + stc);
            bf16* dst = Xlds + str * 32 + stc;
            dst[0] = (bf16)xv.x;
            dst[1] = (bf16)xv.y;
            dst[2] = (bf16)xv.z;
            dst[3] = (bf16)xv.w;
        }
        __syncthreads();

        // A fragment (16x32): lanes 0-15 row=lane K=0..15,
        // lanes 16-31 K=16..31 -> two ds_load_b128 per lane.
        const bf16* xa = Xlds + col * 32 + (hi ? 16 : 0);
        const v16bf a = combine8(ld_v8(xa), ld_v8(xa + 8));

        // B fragments (32x16): contiguous 16 bf16 from transposed weights.
        const size_t woff = (size_t)h * DMODEL + kb + (hi ? 16 : 0);
        const v16bf bwq = ld_v16(Wtq + woff);
        const v16bf bwk = ld_v16(Wtk + woff);
        const v16bf bwv = ld_v16(Wtv + woff);

        accQ = WMMA_BF16(a, bwq, accQ);
        accK = WMMA_BF16(a, bwk, accK);
        accV = WMMA_BF16(a, bwv, accV);
        __syncthreads();                  // protect LDS tile before overwrite
    }

    const float addq = bq[h], addk = bk[h], addv = bv[h];
#pragma unroll
    for (int r = 0; r < 8; ++r) {
        const int m = r + (hi ? 8 : 0);          // C layout: VGPR r -> row m
        Qb[(size_t)(ga + m) * HEAD + h] = (bf16)(accQ[r] + addq);
        Kb[(size_t)(ga + m) * HEAD + h] = (bf16)(accK[r] + addk);
        // V stored transposed: Vt[b][h][s]
        Vt[(size_t)b * HEAD * SEQ + (size_t)h * SEQ + (s0 + m)] =
            (bf16)(accV[r] + addv);
    }
}

// ---------------------------------------------------------------------------
// Phase 2: transposed flash attention.  Each wave owns one 16-query tile of
// one batch; keys processed in chunks of 32.
//   S^T tile  = WMMA(A = K chunk [16 keys x 32 h], B = Q^T [32 h x 16 q])
//   O^T frags = WMMA(A = V^T [16 h x 32 keys],     B = P^T [32 keys x 16 q])
// C layout puts one query per lane -> scalar online-softmax state per lane.
// 8 independent waves per block (private LDS slices; same-wave DS ops are
// in-order on CDNA5, so no barriers needed).
// ---------------------------------------------------------------------------
__global__ void __launch_bounds__(256)
attn_flash_kernel(const bf16* __restrict__ Qb, const bf16* __restrict__ Kb,
                  const bf16* __restrict__ Vt, float* __restrict__ out) {
    __shared__ bf16 Plds[8 * 16 * 32];     // per-wave P^T tile: [query][32 keys]

    const int wid  = threadIdx.x >> 5;
    const int lane = threadIdx.x & 31;
    const int hi   = lane >> 4;
    const int col  = lane & 15;

    const int qtile = blockIdx.x * 8 + wid;
    const int b     = qtile >> 7;          // /128
    const int q0    = (qtile & 127) << 4;

    const bf16* Kbase = Kb + (size_t)(b * SEQ) * HEAD;
    const bf16* Vbase = Vt + (size_t)b * HEAD * SEQ;
    bf16* pl = Plds + wid * (16 * 32);

    // Q^T B-fragments (32 h x 16 queries): lane col owns query q0+col,
    // 8+8 bf16 per h-chunk from the Q row (same k order as the K fragments).
    const bf16* qrow = Qb + (size_t)(b * SEQ + q0 + col) * HEAD + (hi ? 8 : 0);
    const v16bf qb0 = combine8(ld_v8(qrow), ld_v8(qrow + 16));        // h  0..31
    const v16bf qb1 = combine8(ld_v8(qrow + 32), ld_v8(qrow + 48));   // h 32..63

    // V^T A-fragments base: row h = n*16 + col, keys along the row.
    const bf16* vrow = Vbase + (size_t)col * SEQ;

    float m = -INFINITY, l = 0.0f;         // per-lane scalar softmax state
    v8f o0 = {}, o1 = {}, o2 = {}, o3 = {};

    const float scale = 0.125f;            // 1/sqrt(64)

    for (int kb = 0; kb < SEQ; kb += 32) {
        // K A-fragments (16 keys x 32 h): row = key kb + t*16 + col.
        const bf16* kp0 = Kbase + (size_t)(kb + col) * HEAD + (hi ? 8 : 0);
        const bf16* kp1 = Kbase + (size_t)(kb + 16 + col) * HEAD + (hi ? 8 : 0);
        const v16bf ka00 = combine8(ld_v8(kp0),      ld_v8(kp0 + 16)); // h 0..31
        const v16bf ka01 = combine8(ld_v8(kp0 + 32), ld_v8(kp0 + 48)); // h 32..63
        const v16bf ka10 = combine8(ld_v8(kp1),      ld_v8(kp1 + 16));
        const v16bf ka11 = combine8(ld_v8(kp1 + 32), ld_v8(kp1 + 48));

        v8f s0 = {}, s1 = {};              // S^T tiles: rows=keys, col=query
        s0 = WMMA_BF16(ka00, qb0, s0);
        s0 = WMMA_BF16(ka01, qb1, s0);
        s1 = WMMA_BF16(ka10, qb0, s1);
        s1 = WMMA_BF16(ka11, qb1, s1);

        // ---- per-lane online softmax over 32 keys of this lane's query ----
        float mr = -INFINITY;
#pragma unroll
        for (int r = 0; r < 8; ++r) {
            s0[r] *= scale;
            s1[r] *= scale;
            mr = fmaxf(mr, fmaxf(s0[r], s1[r]));
        }
        mr = fmaxf(mr, __shfl_xor(mr, 16));     // merge the two key half-groups
        const float mn = fmaxf(m, mr);
        const float al = __expf(m - mn);
        m = mn;

        float sum = 0.0f;
        v8bf p0v, p1v;
#pragma unroll
        for (int r = 0; r < 8; ++r) {
            const float p0 = __expf(s0[r] - mn);
            const float p1 = __expf(s1[r] - mn);
            sum += p0 + p1;
            p0v[r] = (bf16)p0;
            p1v[r] = (bf16)p1;
        }
        sum += __shfl_xor(sum, 16);
        l = l * al + sum;

#pragma unroll
        for (int r = 0; r < 8; ++r) {
            o0[r] *= al; o1[r] *= al; o2[r] *= al; o3[r] *= al;
        }

        // P^T relayout: [query][key] in LDS; keys (r + hi*8) are consecutive
        // -> two packed ds_store_b128 per lane.
        bf16* pw = pl + col * 32 + hi * 8;
        *(v8bf*)pw        = p0v;            // keys  0..15 region
        *(v8bf*)(pw + 16) = p1v;            // keys 16..31 region

        // P^T B-fragment (32 keys x 16 q): 8+8 keys per lane.
        const bf16* pr = pl + col * 32 + (hi ? 8 : 0);
        const v16bf pb = combine8(ld_v8(pr), ld_v8(pr + 16));

        // V^T A-fragments (16 h x 32 keys): contiguous 8+8 keys per lane.
        const bf16* vp = vrow + kb + (hi ? 8 : 0);
        const v16bf va0 = combine8(ld_v8(vp),                ld_v8(vp + 16));
        const v16bf va1 = combine8(ld_v8(vp + 16 * SEQ),     ld_v8(vp + 16 * SEQ + 16));
        const v16bf va2 = combine8(ld_v8(vp + 32 * SEQ),     ld_v8(vp + 32 * SEQ + 16));
        const v16bf va3 = combine8(ld_v8(vp + 48 * SEQ),     ld_v8(vp + 48 * SEQ + 16));

        o0 = WMMA_BF16(va0, pb, o0);
        o1 = WMMA_BF16(va1, pb, o1);
        o2 = WMMA_BF16(va2, pb, o2);
        o3 = WMMA_BF16(va3, pb, o3);
    }

    // Epilogue: O^T C layout -> each lane holds 8 consecutive h values of its
    // query per fragment; normalize and store as contiguous 32B v8f.
    const float inv = 1.0f / l;
    float* op = out + (size_t)(b * SEQ + q0 + col) * HEAD + hi * 8;
    v8f r0, r1, r2, r3;
#pragma unroll
    for (int r = 0; r < 8; ++r) {
        r0[r] = o0[r] * inv;
        r1[r] = o1[r] * inv;
        r2[r] = o2[r] * inv;
        r3[r] = o3[r] * inv;
    }
    *(v8f*)(op)      = r0;
    *(v8f*)(op + 16) = r1;
    *(v8f*)(op + 32) = r2;
    *(v8f*)(op + 48) = r3;
}

extern "C" void kernel_launch(void* const* d_in, const int* in_sizes, int n_in,
                              void* d_out, int out_size, void* d_ws, size_t ws_size,
                              hipStream_t stream) {
    const float* x  = (const float*)d_in[0];
    const float* Wq = (const float*)d_in[1];
    const float* bq = (const float*)d_in[2];
    const float* Wk = (const float*)d_in[3];
    const float* bk = (const float*)d_in[4];
    const float* Wv = (const float*)d_in[5];
    const float* bv = (const float*)d_in[6];
    float* out = (float*)d_out;

    // Workspace: Q bf16 (2MB) | K bf16 (2MB) | V^T bf16 (2MB) | Wt q/k/v (3x96KB)
    bf16* Qb  = (bf16*)d_ws;
    bf16* Kb  = Qb  + (size_t)BATCH * SEQ * HEAD;
    bf16* Vt  = Kb  + (size_t)BATCH * SEQ * HEAD;
    bf16* Wtq = Vt  + (size_t)BATCH * SEQ * HEAD;
    bf16* Wtk = Wtq + (size_t)DMODEL * HEAD;
    bf16* Wtv = Wtk + (size_t)DMODEL * HEAD;

    attn_wt_transpose_kernel<<<dim3(DMODEL * HEAD / 256), 256, 0, stream>>>(
        Wq, Wk, Wv, Wtq, Wtk, Wtv);

    attn_qkv_proj_kernel<<<dim3(BATCH * SEQ / 16), 128, 0, stream>>>(
        x, Wtq, bq, Wtk, bk, Wtv, bv, Qb, Kb, Vt);

    attn_flash_kernel<<<dim3(BATCH * QTILES_PER_B / 8), 256, 0, stream>>>(
        Qb, Kb, Vt, out);
}
